// BatchedSimpleEIColumns_14448269984073
// MI455X (gfx1250) — matrix operand order repaired and checked by reading, not compiled, not run on hardware
//
#include <hip/hip_runtime.h>
#include <hip/hip_bf16.h>

// Dims: batch=64, n_cols=512, n_e=80, n_i=20, embed=512
#define BB  64
#define CC  512
#define NEE 80
#define NII 20
#define EE  512

typedef __attribute__((ext_vector_type(4)))  __bf16 v4bf;
typedef __attribute__((ext_vector_type(8)))  __bf16 v8bf;
typedef __attribute__((ext_vector_type(16))) __bf16 v16bf;
typedef __attribute__((ext_vector_type(8)))  float  v8f;

// round-to-nearest-even f32 -> bf16
__device__ __forceinline__ __bf16 f2bf(float f) {
  union { float f; unsigned u; } v; v.f = f;
  unsigned r = v.u + 0x7FFFu + ((v.u >> 16) & 1u);
  unsigned short h = (unsigned short)(r >> 16);
  return __builtin_bit_cast(__bf16, h);
}

__device__ __forceinline__ v8f wmma_bf16(v16bf a, v16bf b, v8f c) {
  // (neg_a, A, neg_b, B, c_mod, C, reuse_a, reuse_b)
  return __builtin_amdgcn_wmma_f32_16x16x32_bf16(false, a, false, b, (short)0, c,
                                                 false, false);
}

// A-matrix 16x32 bf16 fragment (ISA 7.12.2): lanes 0-15 row M=lane hold
// K {k0+0..7, k0+16..23}; lanes 16-31 hold K {k0+8..15, k0+24..31}.
__device__ __forceinline__ v16bf ld_fragA(const __bf16* base, int stride,
                                          int m0, int k0, int lane) {
  const __bf16* p = base + (m0 + (lane & 15)) * stride + k0 + ((lane >> 4) << 3);
  v8bf lo = *(const v8bf*)(p);
  v8bf hi = *(const v8bf*)(p + 16);
  v16bf r;
#pragma unroll
  for (int i = 0; i < 8; i++) { r[i] = lo[i]; r[i + 8] = hi[i]; }
  return r;
}

// B-matrix 32x16 bf16 fragment: lane l holds column N=l&15; lanes 0-15 hold
// K {k0+0..7, k0+16..23}, lanes 16-31 hold K {k0+8..15, k0+24..31} (matches A).
__device__ __forceinline__ v16bf ld_fragB(const __bf16* base, int stride,
                                          int n0, int k0, int lane) {
  const __bf16* p = base + (n0 + (lane & 15)) * stride + k0 + ((lane >> 4) << 3);
  v8bf lo = *(const v8bf*)(p);
  v8bf hi = *(const v8bf*)(p + 16);
  v16bf r;
#pragma unroll
  for (int i = 0; i < 8; i++) { r[i] = lo[i]; r[i + 8] = hi[i]; }
  return r;
}

// Cooperative f32->bf16 staging with zero padding (rows beyond rowsValid and
// K beyond kValid zeroed so padded WMMA K-steps contribute nothing).
template <int KPAD>
__device__ __forceinline__ void stage_bf(__bf16* dst, int stride, int rowsPad,
                                         int rowsValid, int kValid,
                                         const float* src, long srcStride,
                                         float sign, int tid) {
  const int total = rowsPad * KPAD;
  for (int i = tid; i < total; i += 256) {
    int r = i / KPAD, k = i - r * KPAD;
    float v = 0.0f;
    if (r < rowsValid && k < kValid) v = sign * src[(long)r * srcStride + k];
    dst[r * stride + k] = f2bf(v);
  }
}

// One wave's excitatory tiles: shared M-tile, 3 N-tiles -> the inner unrolled
// loop issues WMMAs to independent accumulators (no RAW hazard NOP chains).
// All waves run this with full EXEC; no divergence around WMMA.
__device__ __forceinline__ void mma_exc3(v8f* acc, const int* nts, int mt,
                                         const __bf16* A, const __bf16* W,
                                         int stride, int kTot, int lane) {
  for (int k = 0; k < kTot; k += 32) {
    v16bf a = ld_fragA(A, stride, mt * 16, k, lane);
#pragma unroll
    for (int i = 0; i < 3; i++) {
      v16bf b = ld_fragB(W, stride, nts[i] * 16, k, lane);
      acc[i] = wmma_bf16(a, b, acc[i]);
    }
  }
}

__global__ __launch_bounds__(256) void BatchedSimpleEIColumns_kernel(
    const float* __restrict__ thal,           // [B,E]
    const float* __restrict__ thal_inc,       // [B,C,E]
    const float* __restrict__ l23_fb,         // [B,C,NE]
    const float* __restrict__ r_e,            // [B,C,NE]
    const float* __restrict__ r_i,            // [B,C,NI]
    const float* __restrict__ e_v,            // [B,C,NE]
    const float* __restrict__ i_v,            // [B,C,NI]
    const float* __restrict__ input_proj,     // [C,NE,E]
    const float* __restrict__ feedback_proj,  // [C,NE,NE]
    const float* __restrict__ W_ee,           // [C,NE,NE]
    const float* __restrict__ W_ei,           // [C,NI,NE]
    const float* __restrict__ W_ie,           // [C,NE,NI]
    float* __restrict__ out)                  // r_e',r_i',v_e',v_i' concat
{
  // LDS pools (reused across phases): 8704+10880+3328 bf16 = ~46 KB
  __shared__ __align__(16) __bf16 sA[BB * 136];    // activations / A chunks
  __shared__ __align__(16) __bf16 sW[NEE * 136];   // weights / W chunks
  __shared__ __align__(16) __bf16 sW2[32 * 104];   // W_ei (N padded 20->32)

  const int tid  = threadIdx.x;
  const int wave = tid >> 5;
  const int lane = tid & 31;
  const int c    = blockIdx.x;

  // Excitatory 16x16 tiles: t in [0,20), mt = t&3, nt = t>>2.
  // wave w owns t = w, w+8, w+16 -> identical mt per wave. Waves 4-7 only own
  // two unique tiles; they redundantly recompute tile 1 into acc[2] so that
  // every wave executes the exact same (branch-free, full-EXEC) WMMA code.
  const int mt   = wave & 3;   // shared M tile (batch rows mt*16..mt*16+15)
  const int ntb  = wave >> 2;
  const bool has3 = (wave < 4);
  int nts[3];
  nts[0] = ntb;
  nts[1] = ntb + 2;
  nts[2] = has3 ? (ntb + 4) : (ntb + 2);
  const int nStore = has3 ? 3 : 2;
  const int nti = wave >> 2;   // inhibitory N tile (0 or 1; n>=20 discarded)

  v8f accE[3]; v8f accI;
#pragma unroll
  for (int i = 0; i < 8; i++) {
    accI[i] = 0.f; accE[0][i] = 0.f; accE[1][i] = 0.f; accE[2][i] = 0.f;
  }

  // ---- Phase 1a: I_fb = l23_fb @ feedback_proj^T (K = 80 -> pad 96) ----
  stage_bf<96>(sA, 104, BB, BB, NEE, l23_fb + (long)c * NEE, (long)CC * NEE, 1.0f, tid);
  stage_bf<96>(sW, 104, NEE, NEE, NEE, feedback_proj + (long)c * NEE * NEE, NEE, 1.0f, tid);
  __syncthreads();
  mma_exc3(accE, nts, mt, sA, sW, 104, 96, lane);
  __syncthreads();

  // ---- Phase 1b: I_ee = r_e @ W_ee^T  and  I_ei = r_e @ W_ei^T ----
  stage_bf<96>(sA, 104, BB, BB, NEE, r_e + (long)c * NEE, (long)CC * NEE, 1.0f, tid);
  stage_bf<96>(sW, 104, NEE, NEE, NEE, W_ee + (long)c * NEE * NEE, NEE, 1.0f, tid);
  stage_bf<96>(sW2, 104, 32, NII, NEE, W_ei + (long)c * NII * NEE, NEE, 1.0f, tid);
  __syncthreads();
  mma_exc3(accE, nts, mt, sA, sW, 104, 96, lane);
  for (int k = 0; k < 96; k += 32) {
    v16bf a = ld_fragA(sA, 104, mt * 16, k, lane);
    v16bf b = ld_fragB(sW2, 104, nti * 16, k, lane);
    accI = wmma_bf16(a, b, accI);
  }
  __syncthreads();

  // ---- Phase 1c: I_ie = -(r_i @ W_ie^T)  (K = 20 -> pad 32; sign in LDS) ----
  stage_bf<32>(sA, 40, BB, BB, NII, r_i + (long)c * NII, (long)CC * NII, 1.0f, tid);
  stage_bf<32>(sW, 40, NEE, NEE, NII, W_ie + (long)c * NEE * NII, NII, -1.0f, tid);
  __syncthreads();
  mma_exc3(accE, nts, mt, sA, sW, 40, 32, lane);
  __syncthreads();

  // ---- Phase 2: I_ext = (thal + thal_inc) @ input_proj^T, K=512 in 128 chunks ----
  for (int k0 = 0; k0 < EE; k0 += 128) {
    // A chunk: fuse the broadcast add during the f32->bf16 convert
    for (int i = tid; i < BB * 32; i += 256) {
      int b = i >> 5, q = i & 31, e = k0 + (q << 2);
      float4 t4 = *(const float4*)(thal + (long)b * EE + e);
      float4 i4 = *(const float4*)(thal_inc + ((long)b * CC + c) * EE + e);
      v4bf o;
      o[0] = f2bf(t4.x + i4.x); o[1] = f2bf(t4.y + i4.y);
      o[2] = f2bf(t4.z + i4.z); o[3] = f2bf(t4.w + i4.w);
      *(v4bf*)(sA + b * 136 + (q << 2)) = o;
    }
    // W chunk
    for (int i = tid; i < NEE * 32; i += 256) {
      int o = i >> 5, q = i & 31, e = k0 + (q << 2);
      float4 w4 = *(const float4*)(input_proj + ((long)c * NEE + o) * EE + e);
      v4bf v;
      v[0] = f2bf(w4.x); v[1] = f2bf(w4.y); v[2] = f2bf(w4.z); v[3] = f2bf(w4.w);
      *(v4bf*)(sW + o * 136 + (q << 2)) = v;
    }
    // gfx1250 global_prefetch_b8 on next K chunk
    if (k0 + 128 < EE) {
      __builtin_prefetch(input_proj + ((long)c * NEE + (tid % NEE)) * EE + (k0 + 128), 0, 1);
      if (tid < BB)
        __builtin_prefetch(thal_inc + ((long)tid * CC + c) * EE + (k0 + 128), 0, 1);
    }
    __syncthreads();
    mma_exc3(accE, nts, mt, sA, sW, 136, 128, lane);
    __syncthreads();
  }

  // ---- Fused leaky-integration + ReLU + store ----
  // C/D layout: element j of v8f -> row m = mt*16 + j + 8*(lane>=16), col n = lane&15
  const long offRI = (long)BB * CC * NEE;
  const long offVE = offRI + (long)BB * CC * NII;
  const long offVI = offVE + (long)BB * CC * NEE;
  const int  nn = lane & 15;
  const int  mb = mt * 16 + ((lane >> 4) << 3);

  for (int ti = 0; ti < nStore; ti++) {
    const int n = nts[ti] * 16 + nn;
#pragma unroll
    for (int j = 0; j < 8; j++) {
      const int  m   = mb + j;
      const long idx = ((long)m * CC + c) * NEE + n;
      const float ev = e_v[idx];
      const float vn = ev + 0.1f * (accE[ti][j] - ev);
      out[offVE + idx] = vn;              // v_e_new
      out[idx]         = fmaxf(vn, 0.0f); // r_e_new
    }
  }
  {
    const int n = nti * 16 + nn;
    if (n < NII) {
#pragma unroll
      for (int j = 0; j < 8; j++) {
        const int  m   = mb + j;
        const long idx = ((long)m * CC + c) * NII + n;
        const float iv = i_v[idx];
        const float vn = iv + 0.1f * (accI[j] - iv);
        out[offVI + idx] = vn;              // v_i_new
        out[offRI + idx] = fmaxf(vn, 0.0f); // r_i_new
      }
    }
  }
}

extern "C" void kernel_launch(void* const* d_in, const int* in_sizes, int n_in,
                              void* d_out, int out_size, void* d_ws, size_t ws_size,
                              hipStream_t stream) {
  (void)in_sizes; (void)n_in; (void)out_size; (void)d_ws; (void)ws_size;
  const float* thal          = (const float*)d_in[0];
  const float* thal_inc      = (const float*)d_in[1];
  const float* l23_fb        = (const float*)d_in[2];
  const float* r_e           = (const float*)d_in[3];
  const float* r_i           = (const float*)d_in[4];
  const float* e_v           = (const float*)d_in[5];
  const float* i_v           = (const float*)d_in[6];
  const float* input_proj    = (const float*)d_in[7];
  const float* feedback_proj = (const float*)d_in[8];
  const float* W_ee          = (const float*)d_in[9];
  const float* W_ei          = (const float*)d_in[10];
  const float* W_ie          = (const float*)d_in[11];
  float* out = (float*)d_out;

  BatchedSimpleEIColumns_kernel<<<dim3(CC), dim3(256), 0, stream>>>(
      thal, thal_inc, l23_fb, r_e, r_i, e_v, i_v,
      input_proj, feedback_proj, W_ee, W_ei, W_ie, out);
}
